// SupervisedContrastiveLoss_13606456394199
// MI455X (gfx1250) — matrix operand order repaired and checked by reading, not compiled, not run on hardware
//
#include <hip/hip_runtime.h>
#include <hip/hip_bf16.h>

typedef __attribute__((ext_vector_type(16))) _Float16 v16h;
typedef __attribute__((ext_vector_type(8)))  _Float16 v8h;
typedef __attribute__((ext_vector_type(4)))  _Float16 v4h;
typedef __attribute__((ext_vector_type(8)))  float    v8f;

#define NROWS 4096
#define DIM   1024
#define EPSF  1e-10f
#define INV_T 10.0f
#define LDS_STRIDE 72   // halves per LDS tile row (32 data + 40 pad) -> conflict-free b128 reads

// ---------------------------------------------------------------------------
// Async global->LDS copy (CDNA5 ASYNCcnt path). Inline asm: no probe-confirmed
// builtin for these. LDS operand is the low 32 bits of the flat shared address
// (ISA: LDS flat addr low 32 bits == LDS byte offset).
// ---------------------------------------------------------------------------
__device__ __forceinline__ void async_b128_to_lds(void* lds_ptr, const void* gptr) {
    const unsigned lds_off = (unsigned)(size_t)lds_ptr;
    asm volatile("global_load_async_to_lds_b128 %0, %1, off"
                 :: "v"(lds_off), "v"(gptr) : "memory");
}
__device__ __forceinline__ void wait_asynccnt0() {
    asm volatile("s_wait_asynccnt 0x0" ::: "memory");
}

// ---------------------------------------------------------------------------
// Kernel 1: row L2-norm in f32, normalized rows written as f16 (WMMA operand)
// ---------------------------------------------------------------------------
__global__ __launch_bounds__(256) void scl_normalize(const float* __restrict__ pred,
                                                     _Float16* __restrict__ xn) {
    const int row  = blockIdx.x;
    const int t    = threadIdx.x;        // 256 threads, 4 floats each
    const int lane = t & 31;
    const int wid  = t >> 5;

    const float4 v = reinterpret_cast<const float4*>(pred + (size_t)row * DIM)[t];
    float ss = v.x * v.x + v.y * v.y + v.z * v.z + v.w * v.w;
    #pragma unroll
    for (int off = 16; off > 0; off >>= 1) ss += __shfl_xor(ss, off, 32);

    __shared__ float wsum[8];
    if (lane == 0) wsum[wid] = ss;
    __syncthreads();
    float tot = 0.f;
    #pragma unroll
    for (int i = 0; i < 8; ++i) tot += wsum[i];

    const float inv = 1.0f / fmaxf(sqrtf(tot), 1e-8f);   // COS_EPS clamp
    v4h h;
    h[0] = (_Float16)(v.x * inv);
    h[1] = (_Float16)(v.y * inv);
    h[2] = (_Float16)(v.z * inv);
    h[3] = (_Float16)(v.w * inv);
    reinterpret_cast<v4h*>(xn + (size_t)row * DIM)[t] = h;
}

// ---------------------------------------------------------------------------
// Issue one double-buffer stage: A tile (64 rows x 32 halves) + B tile into
// LDS via async B128 copies. 256 16-byte chunks per tile / 128 threads = 2 ea.
// ---------------------------------------------------------------------------
__device__ __forceinline__ void stage_tiles(const _Float16* __restrict__ xn,
                                            int mBase, int nBase, int k0, int t,
                                            _Float16* sa, _Float16* sb) {
    #pragma unroll
    for (int h = 0; h < 2; ++h) {
        const int q   = t + h * 128;      // chunk id 0..255
        const int row = q >> 2;           // tile row 0..63
        const int cib = (q & 3) * 8;      // half-offset within row: 0,8,16,24
        async_b128_to_lds(sa + row * LDS_STRIDE + cib,
                          xn + (size_t)(mBase + row) * DIM + k0 + cib);
        async_b128_to_lds(sb + row * LDS_STRIDE + cib,
                          xn + (size_t)(nBase + row) * DIM + k0 + cib);
    }
}

// ---------------------------------------------------------------------------
// Kernels 2/3: WMMA Gram tiles with async-LDS double buffering.
// PASS==1: per-column negative-mask denominators. PASS==2: masked log(frac).
// Block = 128 threads (4 wave32), 64x64 macro tile; wave w owns rows
// [m0,m0+16) x cols [nBase,nBase+64) as 4 accumulators over K=1024.
// ---------------------------------------------------------------------------
template <int PASS>
__global__ __launch_bounds__(128) void scl_gram(const _Float16* __restrict__ xn,
                                                const int* __restrict__ targ,
                                                float* __restrict__ denom,
                                                float* __restrict__ acc) {
    __shared__ __align__(16) _Float16 SA[2][64 * LDS_STRIDE];
    __shared__ __align__(16) _Float16 SB[2][64 * LDS_STRIDE];
    __shared__ int   tRow[64];
    __shared__ int   tCol[64];
    __shared__ float wred[4];

    const int t     = threadIdx.x;
    const int lane  = t & 31;
    const int wave  = t >> 5;
    const int mBase = blockIdx.x * 64;
    const int nBase = blockIdx.y * 64;

    if (t < 64)       tRow[t]      = targ[mBase + t];
    else              tCol[t - 64] = targ[nBase + (t - 64)];

    // A fragment: lane l holds row (l&15); K chunks at kAoff and kAoff+16 (8 halves)
    const int kAoff = (lane >> 4) * 8;
    // B fragment: lane l holds col (l&15); K chunks at kBoff and kBoff+16 (8 halves)
    const int kBoff = (lane >> 4) * 8;
    const int rSelA = (wave * 16 + (lane & 15)) * LDS_STRIDE;

    v8f cacc[4] = {v8f{}, v8f{}, v8f{}, v8f{}};

    stage_tiles(xn, mBase, nBase, 0, t, SA[0], SB[0]);

    int s = 0;
    for (int k0 = 0; k0 < DIM; k0 += 32, s ^= 1) {
        wait_asynccnt0();       // this wave's async copies for stage s landed
        __syncthreads();        // ...and everyone else's; prior stage reads done
        if (k0 + 32 < DIM)
            stage_tiles(xn, mBase, nBase, k0 + 32, t, SA[s ^ 1], SB[s ^ 1]);

        const _Float16* sa = SA[s];
        const _Float16* sb = SB[s];

        const v8h alo = *(const v8h*)(sa + rSelA + kAoff);
        const v8h ahi = *(const v8h*)(sa + rSelA + kAoff + 16);
        const v16h a = __builtin_shufflevector(alo, ahi,
            0, 1, 2, 3, 4, 5, 6, 7, 8, 9, 10, 11, 12, 13, 14, 15);
        #pragma unroll
        for (int nt = 0; nt < 4; ++nt) {
            const _Float16* br = sb + (nt * 16 + (lane & 15)) * LDS_STRIDE + kBoff;
            const v8h blo = *(const v8h*)(br);
            const v8h bhi = *(const v8h*)(br + 16);
            const v16h b = __builtin_shufflevector(blo, bhi,
                0, 1, 2, 3, 4, 5, 6, 7, 8, 9, 10, 11, 12, 13, 14, 15);
            cacc[nt] = __builtin_amdgcn_wmma_f32_16x16x32_f16(
                false, a, false, b, (short)0, cacc[nt], false, false);
        }
    }

    // C/D layout: lane l -> col (l&15); element r -> row m0 + r + (l>=16 ? 8 : 0)
    const int rOff = wave * 16 + (lane >> 4) * 8;   // local row base within macro tile
    const int jLoc = lane & 15;
    float lossLocal = 0.f;

    #pragma unroll
    for (int nt = 0; nt < 4; ++nt) {
        const int gj = nBase + nt * 16 + jLoc;
        const int tj = tCol[nt * 16 + jLoc];
        if (PASS == 1) {
            float colsum = 0.f;
            #pragma unroll
            for (int r = 0; r < 8; ++r) {
                const float p    = fmaxf(cacc[nt][r], EPSF);
                const float dist = __expf(p * INV_T);
                if (tRow[rOff + r] != tj) colsum += dist;   // negatives (diag excluded)
            }
            colsum += __shfl_xor(colsum, 16, 32);           // fold both lane halves
            if (lane < 16) atomicAdd(&denom[gj], colsum);
        } else {
            const float d    = fmaxf(denom[gj], EPSF);
            const float invD = 1.0f / d;
            #pragma unroll
            for (int r = 0; r < 8; ++r) {
                const int gi = mBase + rOff + r;
                if (gi != gj && tRow[rOff + r] == tj) {     // positives, off-diagonal
                    const float p    = fmaxf(cacc[nt][r], EPSF);
                    const float frac = __expf(p * INV_T) * invD;
                    if (frac >= EPSF) lossLocal += __logf(frac);
                }
            }
        }
    }

    if (PASS == 2) {
        #pragma unroll
        for (int off = 16; off > 0; off >>= 1) lossLocal += __shfl_xor(lossLocal, off, 32);
        if (lane == 0) wred[wave] = lossLocal;
        __syncthreads();
        if (t == 0) atomicAdd(acc, wred[0] + wred[1] + wred[2] + wred[3]);
    }
}

__global__ void scl_finalize(const float* __restrict__ acc, float* __restrict__ out) {
    if (threadIdx.x == 0) out[0] = -acc[0] * (1.0f / 4096.0f);
}

// ---------------------------------------------------------------------------
extern "C" void kernel_launch(void* const* d_in, const int* in_sizes, int n_in,
                              void* d_out, int out_size, void* d_ws, size_t ws_size,
                              hipStream_t stream) {
    const float* pred = (const float*)d_in[0];
    const int*   targ = (const int*)d_in[1];
    float*       out  = (float*)d_out;

    _Float16* xn    = (_Float16*)d_ws;                                    // 8 MB
    float*    denom = (float*)((char*)d_ws + (size_t)NROWS * DIM * 2);    // 16 KB
    float*    acc   = denom + NROWS;                                      // 4 B

    hipMemsetAsync(denom, 0, (NROWS + 1) * sizeof(float), stream);

    scl_normalize<<<NROWS, 256, 0, stream>>>(pred, xn);

    dim3 grid(NROWS / 64, NROWS / 64);
    scl_gram<1><<<grid, 128, 0, stream>>>(xn, targ, denom, acc);
    scl_gram<2><<<grid, 128, 0, stream>>>(xn, targ, denom, acc);

    scl_finalize<<<1, 32, 0, stream>>>(acc, out);
}
